// Donut3_46007689674776
// MI455X (gfx1250) — matrix operand order repaired and checked by reading, not compiled, not run on hardware
//
#include <hip/hip_runtime.h>
#include <math.h>

// ---------------------------------------------------------------------------
// CDNA5 (gfx1250) wave32 WMMA types
// ---------------------------------------------------------------------------
typedef __bf16 bf16;
typedef bf16  v16bf __attribute__((ext_vector_type(16)));
typedef float v8f   __attribute__((ext_vector_type(8)));

union FragAB { v16bf v; unsigned u[8]; };
union FragC  { v8f   v; float    f[8]; };

// f32 -> bf16, round-half-up (one v_add); single element
__device__ __forceinline__ unsigned short f2bf(float f) {
  union { float f; unsigned u; } x; x.f = f;
  return (unsigned short)((x.u + 0x8000u) >> 16);
}
// pack two f32 -> bf16x2 in one dword: 2x v_add + 1x v_perm_b32
__device__ __forceinline__ unsigned f2bf2(float a, float b) {  // lo=a, hi=b
  union { float f; unsigned u; } x, y; x.f = a; y.f = b;
  unsigned ra = x.u + 0x8000u, rb = y.u + 0x8000u;
  // combined {src0,src1} = {rb, ra}; take ra bytes 2,3 then rb bytes 2,3
  return __builtin_amdgcn_perm(rb, ra, 0x07060302u);
}

// 16-bit A-matrix 16x32 per-lane k index (ISA 7.12.2):
// lanes 0-15: V0..3 -> K 0..7, V4..7 -> K 16..23 ; lanes 16-31 shifted by 8
__device__ __forceinline__ int k0A(int j, int lane) {
  return ((j >> 2) << 4) + ((lane >> 4) << 3) + ((j & 3) << 1);
}
// 16-bit B-matrix 32x16 per-lane k index: same order as A (V0..3 -> K 0..7, V4..7 -> K 16..23; lanes 16-31 +8)
__device__ __forceinline__ int k0B(int j, int lane) {
  return ((j >> 2) << 4) + ((lane >> 4) << 3) + ((j & 3) << 1);
}

// ---------------------------------------------------------------------------
// Generic f32-in / f32-out GEMM via bf16 WMMA.  C[M,N] = A[M,K] * B + bias
// btrans==0: B is [K,N] row-major.  btrans==1: B is [N,K] row-major (B^T).
// Requires M%128==0, N%128==0, K%64==0 (true for every call here).
// Block: 256 threads = 8 waves arranged 4(m) x 2(n); wave tile 32x64.
// ---------------------------------------------------------------------------
#define BM 128
#define BN 128
#define BK 64

__global__ __launch_bounds__(256) void gemm_kernel(
    const float* __restrict__ A, const float* __restrict__ B,
    float* __restrict__ C, int M, int N, int K, int btrans,
    const float* __restrict__ bias)
{
  __shared__ unsigned short sA[BM][BK];   // [m][k]  16 KB
  __shared__ unsigned short sB[BN][BK];   // [n][k]  16 KB
  const int tid  = threadIdx.x, lane = tid & 31, wid = tid >> 5;
  const int wm   = (wid & 3) * 32;        // wave row offset inside tile
  const int wn   = (wid >> 2) * 64;       // wave col offset inside tile
  const int bm   = blockIdx.x * BM, bn = blockIdx.y * BN;

  FragC acc[2][4];
#pragma unroll
  for (int ms = 0; ms < 2; ++ms)
#pragma unroll
    for (int ns = 0; ns < 4; ++ns)
#pragma unroll
      for (int e = 0; e < 8; ++e) acc[ms][ns].f[e] = 0.f;

  for (int kt = 0; kt < K; kt += BK) {
    float4 ta[8], tb[8];
    // ---- batched global loads (8x global_load_b128 each, then convert) ----
#pragma unroll
    for (int t = 0; t < 8; ++t) {                        // A: [m][k] tile
      int i = tid + t * 256, r = i >> 4, c4 = (i & 15) * 4;
      ta[t] = *(const float4*)(A + (size_t)(bm + r) * K + kt + c4);
    }
    if (btrans) {                                        // B from [N,K]
#pragma unroll
      for (int t = 0; t < 8; ++t) {
        int i = tid + t * 256, n = i >> 4, c4 = (i & 15) * 4;
        tb[t] = *(const float4*)(B + (size_t)(bn + n) * K + kt + c4);
      }
    } else {                                             // B from [K,N]
#pragma unroll
      for (int t = 0; t < 8; ++t) {
        int i = tid + t * 256, c = i >> 5, n4 = (i & 31) * 4;
        tb[t] = *(const float4*)(B + (size_t)(kt + c) * N + bn + n4);
      }
    }
    // ---- convert + LDS stores ----
#pragma unroll
    for (int t = 0; t < 8; ++t) {
      int i = tid + t * 256, r = i >> 4, c4 = (i & 15) * 4;
      uint2 w; w.x = f2bf2(ta[t].x, ta[t].y); w.y = f2bf2(ta[t].z, ta[t].w);
      *(uint2*)&sA[r][c4] = w;
    }
    if (btrans) {
#pragma unroll
      for (int t = 0; t < 8; ++t) {
        int i = tid + t * 256, n = i >> 4, c4 = (i & 15) * 4;
        uint2 w; w.x = f2bf2(tb[t].x, tb[t].y); w.y = f2bf2(tb[t].z, tb[t].w);
        *(uint2*)&sB[n][c4] = w;
      }
    } else {                                             // transpose to [n][k]
#pragma unroll
      for (int t = 0; t < 8; ++t) {
        int i = tid + t * 256, c = i >> 5, n4 = (i & 31) * 4;
        sB[n4 + 0][c] = f2bf(tb[t].x);
        sB[n4 + 1][c] = f2bf(tb[t].y);
        sB[n4 + 2][c] = f2bf(tb[t].z);
        sB[n4 + 3][c] = f2bf(tb[t].w);
      }
    }
    __syncthreads();

#pragma unroll
    for (int ks = 0; ks < 2; ++ks) {                     // two x32 k-steps
      FragAB af[2], bfm[4];
#pragma unroll
      for (int ms = 0; ms < 2; ++ms) {
        const unsigned short* p = &sA[wm + ms * 16 + (lane & 15)][ks * 32];
#pragma unroll
        for (int j = 0; j < 8; ++j)
          af[ms].u[j] = *(const unsigned*)(p + k0A(j, lane));
      }
#pragma unroll
      for (int ns = 0; ns < 4; ++ns) {
        const unsigned short* p = &sB[wn + ns * 16 + (lane & 15)][ks * 32];
#pragma unroll
        for (int j = 0; j < 8; ++j)
          bfm[ns].u[j] = *(const unsigned*)(p + k0B(j, lane));
      }
#pragma unroll
      for (int ms = 0; ms < 2; ++ms)
#pragma unroll
        for (int ns = 0; ns < 4; ++ns)
          acc[ms][ns].v = __builtin_amdgcn_wmma_f32_16x16x32_bf16(
              false, af[ms].v, false, bfm[ns].v, (short)0, acc[ms][ns].v,
              false, false);
    }
    __syncthreads();
  }

#pragma unroll
  for (int ms = 0; ms < 2; ++ms)
#pragma unroll
    for (int ns = 0; ns < 4; ++ns)
#pragma unroll
      for (int j = 0; j < 8; ++j) {
        int row = bm + wm + ms * 16 + j + ((lane >> 4) << 3);
        int col = bn + wn + ns * 16 + (lane & 15);
        float v = acc[ms][ns].f[j];
        if (bias != nullptr) v += bias[col];
        C[(size_t)row * N + col] = v;
      }
}

// ---------------------------------------------------------------------------
// Fused flash-style causal attention with analytic parabolic-cycloidal bias.
// grid = (N/64, B*G*HG), block = 128 (4 waves).  Wave owns 16 query rows.
// q:[B,N,512] (g,u,r)  k:[B,N,256] (g,r)  v:[B,N,512] (g,d)  o:[B,N,1024]
// ---------------------------------------------------------------------------
#define NSEQ 1024

__global__ __launch_bounds__(128) void attn_kernel(
    const float* __restrict__ q, const float* __restrict__ k,
    const float* __restrict__ v, float* __restrict__ o,
    const float* __restrict__ pb_a_p, const float* __restrict__ pb_w_p)
{
  __shared__ unsigned short sQ[64][64];    // [qrow][r]   8 KB
  __shared__ unsigned short sK[64][64];    // [key][r]    8 KB (B-layout [n][k])
  __shared__ unsigned short sVt[128][64];  // [d][key]   16 KB (B-layout [n][k])
  __shared__ unsigned short sP[4][16][64]; // per-wave P  8 KB

  const int tid = threadIdx.x, lane = tid & 31, wid = tid >> 5;
  const int qt = blockIdx.x;
  const int hy = blockIdx.y;                 // b*8 + g*2 + u
  const int b = hy >> 3, g = (hy >> 1) & 3, u = hy & 1;
  const float pba = pb_a_p[0];
  const float pw0 = pb_w_p[0], pw1 = pb_w_p[1], pw2 = pb_w_p[2];

  // load Q tile (64x64): batch 8 loads then convert
  {
    float4 tq[8];
#pragma unroll
    for (int t = 0; t < 8; ++t) {
      int i = tid + t * 128, r = i >> 4, c4 = (i & 15) * 4;
      tq[t] = *(const float4*)(q + (size_t)(b * NSEQ + qt * 64 + r) * 512
                                 + (g * 2 + u) * 64 + c4);
    }
#pragma unroll
    for (int t = 0; t < 8; ++t) {
      int i = tid + t * 128, r = i >> 4, c4 = (i & 15) * 4;
      uint2 w; w.x = f2bf2(tq[t].x, tq[t].y); w.y = f2bf2(tq[t].z, tq[t].w);
      *(uint2*)&sQ[r][c4] = w;
    }
  }
  __syncthreads();

  FragAB aQ[2];
#pragma unroll
  for (int ks = 0; ks < 2; ++ks) {
    const unsigned short* p = &sQ[wid * 16 + (lane & 15)][ks * 32];
#pragma unroll
    for (int j = 0; j < 8; ++j) aQ[ks].u[j] = *(const unsigned*)(p + k0A(j, lane));
  }

  FragC oacc[8];
#pragma unroll
  for (int dt = 0; dt < 8; ++dt)
#pragma unroll
    for (int j = 0; j < 8; ++j) oacc[dt].f[j] = 0.f;
  float rowm[8], rowl[8];
#pragma unroll
  for (int j = 0; j < 8; ++j) { rowm[j] = -1e30f; rowl[j] = 0.f; }

  for (int kt = 0; kt <= qt; ++kt) {
    // K tile (64x64): batched
    {
      float4 tk[8];
#pragma unroll
      for (int t = 0; t < 8; ++t) {
        int i = tid + t * 128, r = i >> 4, c4 = (i & 15) * 4;
        tk[t] = *(const float4*)(k + (size_t)(b * NSEQ + kt * 64 + r) * 256
                                   + g * 64 + c4);
      }
#pragma unroll
      for (int t = 0; t < 8; ++t) {
        int i = tid + t * 128, r = i >> 4, c4 = (i & 15) * 4;
        uint2 w; w.x = f2bf2(tk[t].x, tk[t].y); w.y = f2bf2(tk[t].z, tk[t].w);
        *(uint2*)&sK[r][c4] = w;
      }
    }
    // V tile (64 keys x 128 d), store transposed [d][key]; two batches of 8
#pragma unroll
    for (int half = 0; half < 2; ++half) {
      float4 tv[8];
#pragma unroll
      for (int t = 0; t < 8; ++t) {
        int i = tid + (half * 8 + t) * 128, r = i >> 5, c4 = (i & 31) * 4;
        tv[t] = *(const float4*)(v + (size_t)(b * NSEQ + kt * 64 + r) * 512
                                   + g * 128 + c4);
      }
#pragma unroll
      for (int t = 0; t < 8; ++t) {
        int i = tid + (half * 8 + t) * 128, r = i >> 5, c4 = (i & 31) * 4;
        sVt[c4 + 0][r] = f2bf(tv[t].x);
        sVt[c4 + 1][r] = f2bf(tv[t].y);
        sVt[c4 + 2][r] = f2bf(tv[t].z);
        sVt[c4 + 3][r] = f2bf(tv[t].w);
      }
    }
    __syncthreads();

    FragC sacc[4];
#pragma unroll
    for (int nt = 0; nt < 4; ++nt) {
#pragma unroll
      for (int j = 0; j < 8; ++j) sacc[nt].f[j] = 0.f;
#pragma unroll
      for (int ks = 0; ks < 2; ++ks) {
        FragAB bK;
        const unsigned short* p = &sK[nt * 16 + (lane & 15)][ks * 32];
#pragma unroll
        for (int j = 0; j < 8; ++j) bK.u[j] = *(const unsigned*)(p + k0B(j, lane));
        sacc[nt].v = __builtin_amdgcn_wmma_f32_16x16x32_bf16(
            false, aQ[ks].v, false, bK.v, (short)0, sacc[nt].v, false, false);
      }
    }

    // bias + causal mask + online softmax (rows lane-local: m=j+8*(lane>>4))
    float tmax[8];
#pragma unroll
    for (int j = 0; j < 8; ++j) tmax[j] = -1e30f;
    const int qrow_base = qt * 64 + wid * 16 + ((lane >> 4) << 3);
    const int kcol_base = kt * 64 + (lane & 15);
#pragma unroll
    for (int nt = 0; nt < 4; ++nt)
#pragma unroll
      for (int j = 0; j < 8; ++j) {
        int qr = qrow_base + j, kc = kcol_base + nt * 16;
        float dn = fabsf((float)(qr - kc)) * (1.0f / NSEQ);
        float bs = -pba * dn * dn + pw0 * __sinf(3.14159265f * dn)
                 + pw1 * __sinf(6.28318531f * dn) + pw2 * __sinf(9.42477796f * dn);
        float s = sacc[nt].f[j] * 0.125f + bs;
        if (kc > qr) s = -1e9f;
        sacc[nt].f[j] = s;
        tmax[j] = fmaxf(tmax[j], s);
      }
#pragma unroll
    for (int j = 0; j < 8; ++j)
      for (int m = 8; m >= 1; m >>= 1)
        tmax[j] = fmaxf(tmax[j], __shfl_xor(tmax[j], m, 16));

    float scale[8], psum[8];
#pragma unroll
    for (int j = 0; j < 8; ++j) {
      float nm = fmaxf(rowm[j], tmax[j]);
      scale[j] = __expf(rowm[j] - nm);
      rowm[j] = nm;
      psum[j] = 0.f;
    }
#pragma unroll
    for (int nt = 0; nt < 4; ++nt)
#pragma unroll
      for (int j = 0; j < 8; ++j) {
        float p = __expf(sacc[nt].f[j] - rowm[j]);
        sacc[nt].f[j] = p;
        psum[j] += p;
      }
#pragma unroll
    for (int j = 0; j < 8; ++j) {
      for (int m = 8; m >= 1; m >>= 1) psum[j] += __shfl_xor(psum[j], m, 16);
      rowl[j] = rowl[j] * scale[j] + psum[j];
    }
#pragma unroll
    for (int dt = 0; dt < 8; ++dt)
#pragma unroll
      for (int j = 0; j < 8; ++j) oacc[dt].f[j] *= scale[j];

    // C-frag -> A-frag relayout of P through LDS
#pragma unroll
    for (int nt = 0; nt < 4; ++nt)
#pragma unroll
      for (int j = 0; j < 8; ++j)
        sP[wid][j + ((lane >> 4) << 3)][nt * 16 + (lane & 15)] = f2bf(sacc[nt].f[j]);
    __syncthreads();

    FragAB aP[2];
#pragma unroll
    for (int ks = 0; ks < 2; ++ks) {
      const unsigned short* p = &sP[wid][lane & 15][ks * 32];
#pragma unroll
      for (int j = 0; j < 8; ++j) aP[ks].u[j] = *(const unsigned*)(p + k0A(j, lane));
    }
#pragma unroll
    for (int dt = 0; dt < 8; ++dt)
#pragma unroll
      for (int ks = 0; ks < 2; ++ks) {
        FragAB bV;
        const unsigned short* p = &sVt[dt * 16 + (lane & 15)][ks * 32];
#pragma unroll
        for (int j = 0; j < 8; ++j) bV.u[j] = *(const unsigned*)(p + k0B(j, lane));
        oacc[dt].v = __builtin_amdgcn_wmma_f32_16x16x32_bf16(
            false, aP[ks].v, false, bV.v, (short)0, oacc[dt].v, false, false);
      }
    __syncthreads();
  }

#pragma unroll
  for (int dt = 0; dt < 8; ++dt)
#pragma unroll
    for (int j = 0; j < 8; ++j) {
      int qr = qt * 64 + wid * 16 + j + ((lane >> 4) << 3);
      int d  = (g * 2 + u) * 128 + dt * 16 + (lane & 15);
      o[(size_t)(b * NSEQ + qr) * 1024 + d] = oacc[dt].f[j] / rowl[j];
    }
}

// ---------------------------------------------------------------------------
// LayerNorm (population variance, eps=1e-5). One block per row, float4 I/O.
// ---------------------------------------------------------------------------
__global__ __launch_bounds__(256) void ln_kernel(
    const float* __restrict__ x, const float* __restrict__ g,
    const float* __restrict__ b, float* __restrict__ out, int D_)
{
  __shared__ float red[256];
  const int row = blockIdx.x, tid = threadIdx.x;
  const float4* xr = (const float4*)(x + (size_t)row * D_);
  const int D4 = D_ >> 2;
  float s1 = 0.f, s2 = 0.f;
  for (int i = tid; i < D4; i += 256) {
    float4 v = xr[i];
    s1 += v.x + v.y + v.z + v.w;
    s2 += v.x * v.x + v.y * v.y + v.z * v.z + v.w * v.w;
  }
  red[tid] = s1; __syncthreads();
  for (int off = 128; off > 0; off >>= 1) { if (tid < off) red[tid] += red[tid + off]; __syncthreads(); }
  float mean = red[0] / D_; __syncthreads();
  red[tid] = s2; __syncthreads();
  for (int off = 128; off > 0; off >>= 1) { if (tid < off) red[tid] += red[tid + off]; __syncthreads(); }
  float var = red[0] / D_ - mean * mean;
  float inv = rsqrtf(var + 1e-5f);
  const float4* g4 = (const float4*)g;
  const float4* b4 = (const float4*)b;
  float4* o4 = (float4*)(out + (size_t)row * D_);
  for (int i = tid; i < D4; i += 256) {
    float4 v = xr[i], gg = g4[i], bb = b4[i];
    float4 r;
    r.x = (v.x - mean) * inv * gg.x + bb.x;
    r.y = (v.y - mean) * inv * gg.y + bb.y;
    r.z = (v.z - mean) * inv * gg.z + bb.z;
    r.w = (v.w - mean) * inv * gg.w + bb.w;
    o4[i] = r;
  }
}

// ---------------------------------------------------------------------------
// Small helper kernels (float4 where applicable)
// ---------------------------------------------------------------------------
__global__ void embed_kernel(const int* __restrict__ tok,
                             const float* __restrict__ emb, float* __restrict__ x) {
  const int row = blockIdx.x;
  const float4* src = (const float4*)(emb + (size_t)tok[row] * 1024);
  float4* dst = (float4*)(x + (size_t)row * 1024);
  for (int i = threadIdx.x; i < 256; i += blockDim.x) dst[i] = src[i];
}

__global__ void wsum_kernel(const float* __restrict__ Wpc, const float* __restrict__ cpc,
                            const float* __restrict__ alpha, float* __restrict__ Wsum) {
  const int i = blockIdx.x * 256 + threadIdx.x;
  const int DD = 1024 * 1024;
  if (i < DD)
    Wsum[i] = alpha[0] * (cpc[0] * Wpc[i] + cpc[1] * Wpc[DD + i] + cpc[2] * Wpc[2 * DD + i]);
}

__global__ void add_kernel(float* __restrict__ x, const float* __restrict__ t, int n4) {
  int i = blockIdx.x * 256 + threadIdx.x;
  if (i < n4) {
    float4 a = ((const float4*)x)[i], b = ((const float4*)t)[i];
    a.x += b.x; a.y += b.y; a.z += b.z; a.w += b.w;
    ((float4*)x)[i] = a;
  }
}

__global__ void addtanh_kernel(float* __restrict__ x, const float* __restrict__ t, int n4) {
  int i = blockIdx.x * 256 + threadIdx.x;
  if (i < n4) {
    float4 a = ((const float4*)x)[i], b = ((const float4*)t)[i];
    a.x += 0.02f * tanhf(b.x); a.y += 0.02f * tanhf(b.y);
    a.z += 0.02f * tanhf(b.z); a.w += 0.02f * tanhf(b.w);
    ((float4*)x)[i] = a;
  }
}

__global__ void combine_kernel(float* __restrict__ x, const float* __restrict__ a,
                               const float* __restrict__ s, const float* __restrict__ r,
                               const float* __restrict__ rs, int l, int n4) {
  int i = blockIdx.x * 256 + threadIdx.x;
  if (i < n4) {
    float c = rs[l];
    float4 xv = ((const float4*)x)[i], av = ((const float4*)a)[i];
    float4 sv = ((const float4*)s)[i], rv = ((const float4*)r)[i];
    xv.x += c * (av.x + sv.x + rv.x);
    xv.y += c * (av.y + sv.y + rv.y);
    xv.z += c * (av.z + sv.z + rv.z);
    xv.w += c * (av.w + sv.w + rv.w);
    ((float4*)x)[i] = xv;
  }
}

// diagonal SSM scan: h_n = sigmoid(a_logit)*h_{n-1} + u_n, per (b,s) channel
__global__ void ssm_scan_kernel(const float* __restrict__ u,
                                const float* __restrict__ alog,
                                float* __restrict__ hs) {
  int t = blockIdx.x * blockDim.x + threadIdx.x;
  if (t >= 4 * 128) return;
  int b = t >> 7, s = t & 127;
  float a = 1.f / (1.f + __expf(-alog[s]));
  float h = 0.f;
  const float* ub = u + (size_t)b * 1024 * 128 + s;
  float* hb = hs + (size_t)b * 1024 * 128 + s;
  for (int n = 0; n < 1024; ++n) {
    h = a * h + ub[(size_t)n * 128];
    hb[(size_t)n * 128] = h;
  }
}

// gated RNN scan: one block per batch, LDS-resident hidden state (R=256)
__global__ __launch_bounds__(256) void rnn_scan_kernel(
    const float* __restrict__ xz, const float* __restrict__ xh,
    const float* __restrict__ Uz, const float* __restrict__ Uh,
    float* __restrict__ hr) {
  __shared__ float h[256];
  const int b = blockIdx.x, r = threadIdx.x;
  h[r] = 0.f; __syncthreads();
  for (int n = 0; n < 1024; ++n) {
    size_t base = (size_t)(b * 1024 + n) * 256;
    float az = xz[base + r], ah = xh[base + r];
    for (int kk = 0; kk < 256; ++kk) {
      float hv = h[kk];
      az += hv * Uz[kk * 256 + r];
      ah += hv * Uh[kk * 256 + r];
    }
    float z  = 1.f / (1.f + __expf(-az));
    float hn = (1.f - z) * h[r] + z * tanhf(ah);
    __syncthreads();
    h[r] = hn;
    hr[base + r] = hn;
    __syncthreads();
  }
}

// ---------------------------------------------------------------------------
// Orchestration
// ---------------------------------------------------------------------------
extern "C" void kernel_launch(void* const* d_in, const int* in_sizes, int n_in,
                              void* d_out, int out_size, void* d_ws, size_t ws_size,
                              hipStream_t stream) {
  (void)in_sizes; (void)n_in; (void)out_size; (void)ws_size;
  const int*   tokens   = (const int*)d_in[0];
  const float* embed    = (const float*)d_in[1];
  const float* pb_a     = (const float*)d_in[2];
  const float* pb_w     = (const float*)d_in[3];
  const float* pc_alpha = (const float*)d_in[4];
  const float* W_pc     = (const float*)d_in[5];
  const float* c_pc     = (const float*)d_in[6];
  const float* W_logic  = (const float*)d_in[7];
  const float* ln_g     = (const float*)d_in[8];
  const float* ln_b     = (const float*)d_in[9];
  const float* Wq       = (const float*)d_in[10];
  const float* Wk       = (const float*)d_in[11];
  const float* Wv       = (const float*)d_in[12];
  const float* Wo       = (const float*)d_in[13];
  const float* a_logit  = (const float*)d_in[14];
  const float* Win_ssm  = (const float*)d_in[15];
  const float* Wout_ssm = (const float*)d_in[16];
  const float* Wz       = (const float*)d_in[17];
  const float* Uz       = (const float*)d_in[18];
  const float* Wh       = (const float*)d_in[19];
  const float* Uh       = (const float*)d_in[20];
  const float* Wout_rnn = (const float*)d_in[21];
  const float* res_sc   = (const float*)d_in[22];
  const float* fg       = (const float*)d_in[23];
  const float* fb       = (const float*)d_in[24];
  const float* out_b    = (const float*)d_in[25];
  float* out = (float*)d_out;

  const int D = 1024, M = 4 * 1024, L = 6, V = 32000;

  float* ws  = (float*)d_ws;
  float* x    = ws; ws += (size_t)M * D;
  float* hbuf = ws; ws += (size_t)M * D;
  float* t0   = ws; ws += (size_t)M * D;
  float* attn = ws; ws += (size_t)M * D;
  float* yssm = ws; ws += (size_t)M * D;
  float* yrnn = ws; ws += (size_t)M * D;
  float* qb   = ws; ws += (size_t)M * 512;
  float* kb   = ws; ws += (size_t)M * 256;
  float* vb   = ws; ws += (size_t)M * 512;
  float* ob   = ws; ws += (size_t)M * 1024;
  float* ub   = ws; ws += (size_t)M * 128;
  float* hsb  = ws; ws += (size_t)M * 128;
  float* xzb  = ws; ws += (size_t)M * 256;
  float* xhb  = ws; ws += (size_t)M * 256;
  float* hrb  = ws; ws += (size_t)M * 256;
  float* Wsum = ws; ws += (size_t)D * D;

  const int n4 = (M * D) / 4;

  // embedding + PC projection (K modes folded into one D x D weight) + logic bias
  embed_kernel<<<M, 256, 0, stream>>>(tokens, embed, x);
  wsum_kernel<<<(D * D + 255) / 256, 256, 0, stream>>>(W_pc, c_pc, pc_alpha, Wsum);
  gemm_kernel<<<dim3(M / 128, D / 128), 256, 0, stream>>>(x, Wsum, t0, M, D, D, 0, nullptr);
  add_kernel<<<(n4 + 255) / 256, 256, 0, stream>>>(x, t0, n4);
  gemm_kernel<<<dim3(M / 128, D / 128), 256, 0, stream>>>(x, W_logic, t0, M, D, D, 0, nullptr);
  addtanh_kernel<<<(n4 + 255) / 256, 256, 0, stream>>>(x, t0, n4);

  for (int l = 0; l < L; ++l) {
    ln_kernel<<<M, 256, 0, stream>>>(x, ln_g + l * D, ln_b + l * D, hbuf, D);
    // attention branch
    gemm_kernel<<<dim3(M / 128, 4), 256, 0, stream>>>(hbuf, Wq + (size_t)l * D * 512, qb, M, 512, D, 0, nullptr);
    gemm_kernel<<<dim3(M / 128, 2), 256, 0, stream>>>(hbuf, Wk + (size_t)l * D * 256, kb, M, 256, D, 0, nullptr);
    gemm_kernel<<<dim3(M / 128, 4), 256, 0, stream>>>(hbuf, Wv + (size_t)l * D * 512, vb, M, 512, D, 0, nullptr);
    attn_kernel<<<dim3(1024 / 64, 32), 128, 0, stream>>>(qb, kb, vb, ob, pb_a, pb_w);
    gemm_kernel<<<dim3(M / 128, D / 128), 256, 0, stream>>>(ob, Wo + (size_t)l * 1024 * D, attn, M, D, 1024, 0, nullptr);
    // SSM branch
    gemm_kernel<<<dim3(M / 128, 1), 256, 0, stream>>>(x, Win_ssm + (size_t)l * D * 128, ub, M, 128, D, 0, nullptr);
    ssm_scan_kernel<<<2, 256, 0, stream>>>(ub, a_logit + l * 128, hsb);
    gemm_kernel<<<dim3(M / 128, D / 128), 256, 0, stream>>>(hsb, Wout_ssm + (size_t)l * 128 * D, yssm, M, D, 128, 0, nullptr);
    // gated RNN branch
    gemm_kernel<<<dim3(M / 128, 2), 256, 0, stream>>>(x, Wz + (size_t)l * D * 256, xzb, M, 256, D, 0, nullptr);
    gemm_kernel<<<dim3(M / 128, 2), 256, 0, stream>>>(x, Wh + (size_t)l * D * 256, xhb, M, 256, D, 0, nullptr);
    rnn_scan_kernel<<<4, 256, 0, stream>>>(xzb, xhb, Uz + (size_t)l * 256 * 256,
                                           Uh + (size_t)l * 256 * 256, hrb);
    gemm_kernel<<<dim3(M / 128, D / 128), 256, 0, stream>>>(hrb, Wout_rnn + (size_t)l * 256 * D, yrnn, M, D, 256, 0, nullptr);
    // residual
    combine_kernel<<<(n4 + 255) / 256, 256, 0, stream>>>(x, attn, yssm, yrnn, res_sc, l, n4);
  }

  // tied output head: LN then x_n @ embed^T + out_b   (B^T path, bias epilogue)
  ln_kernel<<<M, 256, 0, stream>>>(x, fg, fb, hbuf, D);
  gemm_kernel<<<dim3(M / 128, V / 128), 256, 0, stream>>>(hbuf, embed, out, M, V, D, 1, out_b);
}
